// EAGAttention_57664230916948
// MI455X (gfx1250) — matrix operand order repaired and checked by reading, not compiled, not run on hardware
//
#include <hip/hip_runtime.h>

#ifndef __has_builtin
#define __has_builtin(x) 0
#endif

#define N_DIM 4096
#define C_DIM 256
#define B_DIM 2

typedef __attribute__((ext_vector_type(16))) __bf16   bf16x16;
typedef __attribute__((ext_vector_type(8)))  float    f32x8;
typedef __attribute__((ext_vector_type(4)))  unsigned u32x4;
typedef __attribute__((ext_vector_type(4)))  int      i32x4;

union Frag16 { bf16x16 v; u32x4 q[2]; };

#if defined(__HIP_DEVICE_COMPILE__)
#if __has_builtin(__builtin_amdgcn_global_load_async_to_lds_b128)
#define HAVE_ASYNC_COPY 1
#else
#define HAVE_ASYNC_COPY 0
#endif
#else
#define HAVE_ASYNC_COPY 0
#endif

#if HAVE_ASYNC_COPY
typedef __attribute__((address_space(1))) i32x4 glob_i32x4;
typedef __attribute__((address_space(3))) i32x4 lds_i32x4;
#endif

__device__ __forceinline__ void cp16(const __bf16* g, __bf16* l) {
#if HAVE_ASYNC_COPY
  __builtin_amdgcn_global_load_async_to_lds_b128(
      (glob_i32x4*)g, (lds_i32x4*)l, 0, 0);
#else
  *(u32x4*)l = *(const u32x4*)g;
#endif
}

// Wait until ASYNCcnt <= N (async loads complete in order, so N=8 means the
// oldest group of 8 per-thread loads has landed while 8 newer stay in flight).
template <int N>
__device__ __forceinline__ void wait_async() {
#if HAVE_ASYNC_COPY
#if __has_builtin(__builtin_amdgcn_s_wait_asynccnt)
  __builtin_amdgcn_s_wait_asynccnt(N);
#else
  asm volatile("s_wait_asynccnt %0" ::"i"(N) : "memory");
#endif
#endif
}

// ---------------------------------------------------------------------------
// Kernel 1: convert the three weight matrices f32 -> bf16 (row-major [d][c]).
// ---------------------------------------------------------------------------
__global__ __launch_bounds__(256) void wcvt_kernel(
    const float* __restrict__ Wq, const float* __restrict__ Wk,
    const float* __restrict__ Wv, __bf16* wq, __bf16* wk, __bf16* wv) {
  int i = blockIdx.x * 256 + threadIdx.x;      // 3 * 65536 threads
  int mi = i >> 16;
  int j  = i & 65535;
  const float* s = (mi == 0) ? Wq : (mi == 1) ? Wk : Wv;
  __bf16*      d = (mi == 0) ? wq : (mi == 1) ? wk : wv;
  d[j] = (__bf16)s[j];
}

// ---------------------------------------------------------------------------
// Kernel 2: QKV projection, bf16 WMMA with f32 accumulate, bias folded in C.
// Block = 256 threads (8 waves), 64 rows per block. grid = B*N/64 = 128.
// q,k stored row-major bf16 [b][n][c]; v stored channel-major [b][c][n].
// ---------------------------------------------------------------------------
#define LDA 264  // padded LDS stride (bf16 elems) for the nodes tile

__global__ __launch_bounds__(256) void qkv_kernel(
    const float* __restrict__ nodes,
    const __bf16* __restrict__ wq, const __bf16* __restrict__ wk,
    const __bf16* __restrict__ wv,
    const float* __restrict__ bq, const float* __restrict__ bk,
    const float* __restrict__ bv,
    __bf16* __restrict__ qb, __bf16* __restrict__ kb, __bf16* __restrict__ vT) {
  __shared__ __bf16 sn[64 * LDA];  // 33.8 KB

  const int tid  = threadIdx.x;
  const int bi   = blockIdx.x;
  const int b    = bi >> 6;
  const int row0 = (bi & 63) << 6;

  // Stage 64x256 nodes tile, f32 -> bf16, into padded LDS.
  {
    const int row = tid >> 2, seg = (tid & 3) * 64;
    const float* src = nodes + ((size_t)(b * N_DIM + row0 + row)) * C_DIM + seg;
    __bf16* dst = &sn[row * LDA + seg];
#pragma unroll
    for (int i = 0; i < 64; i += 2) {
      union { __bf16 h[2]; unsigned u; } pk2;
      pk2.h[0] = (__bf16)src[i];
      pk2.h[1] = (__bf16)src[i + 1];
      *(unsigned*)&dst[i] = pk2.u;
    }
  }
  __syncthreads();

  const int w = tid >> 5, l = tid & 31, m = l & 15, h = l >> 4;
  const int rowsub = w & 3, colhalf = w >> 2;

  // A-fragments (16 rows x 256 ch) for this wave's row sub-tile: 8 frags.
  Frag16 a[8];
  const int arow = (rowsub * 16 + m) * LDA;
#pragma unroll
  for (int kk = 0; kk < 8; ++kk) {
    a[kk].q[0] = *(const u32x4*)&sn[arow + kk * 32 + 16 * h];
    a[kk].q[1] = *(const u32x4*)&sn[arow + kk * 32 + 16 * h + 8];
  }

  for (int p = 0; p < 3; ++p) {
    const __bf16* W    = (p == 0) ? wq : (p == 1) ? wk : wv;
    const float*  bias = (p == 0) ? bq : (p == 1) ? bk : bv;
#pragma unroll
    for (int c8 = 0; c8 < 8; ++c8) {
      const int coltile = colhalf * 8 + c8;
      const int d       = coltile * 16 + m;     // output channel (B col)
      const float bv_   = bias[d];
      f32x8 acc;
#pragma unroll
      for (int i = 0; i < 8; ++i) acc[i] = bv_; // bias in accumulator
      const __bf16* wrow = W + d * C_DIM;       // B[ch,d] = W[d,ch]
#pragma unroll
      for (int kk = 0; kk < 8; ++kk) {
        Frag16 bf;
        bf.q[0] = *(const u32x4*)(wrow + kk * 32 + 16 * h);
        bf.q[1] = *(const u32x4*)(wrow + kk * 32 + 16 * h + 8);
        acc = __builtin_amdgcn_wmma_f32_16x16x32_bf16(
            false, a[kk].v, false, bf.v, (short)0, acc, false, false);
      }
      if (p < 2) {  // q, k: row-major bf16
        __bf16* out  = (p == 0) ? qb : kb;
        size_t  base = ((size_t)(b * N_DIM + row0 + rowsub * 16 + 8 * h)) * C_DIM + d;
#pragma unroll
        for (int r = 0; r < 8; ++r) out[base + (size_t)r * C_DIM] = (__bf16)acc[r];
      } else {      // v: channel-major [b][c][n], 8 consecutive rows -> one b128
        union { __bf16 hh[8]; u32x4 q; } pk;
#pragma unroll
        for (int r = 0; r < 8; ++r) pk.hh[r] = (__bf16)acc[r];
        size_t off = ((size_t)(b * C_DIM + d)) * N_DIM + row0 + rowsub * 16 + 8 * h;
        *(u32x4*)(vT + off) = pk.q;
      }
    }
  }
}

// ---------------------------------------------------------------------------
// Kernel 3: single-pass attention with double-buffered async K/V staging.
// Block = 256 threads (8 waves), each wave owns a 16-row strip (128 rows /
// block), iterating 128 column-steps of 32 columns.
// Fixed-max softmax: p = exp(clip(qk/16) - 10); e[l] cancels in softmax.
// grid = B*N/128 = 64.
// ---------------------------------------------------------------------------
#define KLD 264  // k tile LDS stride (bf16): 32 rows x 256 ch (+pad)
#define VLD 48   // vT tile LDS stride (bf16): 256 ch x 32 rows (+pad)
#define PLD 40   // P transpose buffer stride (bf16): 16 x 32 (+pad)
#define NSTEP 128

// Issue the 8 per-thread async copies staging one 32-column K/V tile.
__device__ __forceinline__ void stage_kv(const __bf16* __restrict__ kb,
                                         const __bf16* __restrict__ vT,
                                         int b, int col0, int tid,
                                         __bf16* skb, __bf16* svb) {
  const int kr = tid >> 3, kc = (tid & 7) * 32;
  const __bf16* ks = kb + ((size_t)(b * N_DIM + col0 + kr)) * C_DIM + kc;
  __bf16* kd = skb + kr * KLD + kc;
  cp16(ks, kd); cp16(ks + 8, kd + 8); cp16(ks + 16, kd + 16); cp16(ks + 24, kd + 24);
  const __bf16* vs = vT + ((size_t)(b * C_DIM + tid)) * N_DIM + col0;
  __bf16* vd = svb + tid * VLD;
  cp16(vs, vd); cp16(vs + 8, vd + 8); cp16(vs + 16, vd + 16); cp16(vs + 24, vd + 24);
}

__global__ __launch_bounds__(256) void attn_kernel(
    const __bf16* __restrict__ qb, const __bf16* __restrict__ kb,
    const __bf16* __restrict__ vT,
    const float* __restrict__ g, const float* __restrict__ e,
    float* __restrict__ Vout, float* __restrict__ Hout) {
  __shared__ __bf16 sk[2][32 * KLD];     // 2 x 16.9 KB
  __shared__ __bf16 sv[2][256 * VLD];    // 2 x 24.6 KB
  __shared__ __bf16 sp[8 * 16 * PLD];    // 10.2 KB (per-wave P buffers)

  const int tid = threadIdx.x, w = tid >> 5, l = tid & 31, m = l & 15, h = l >> 4;
  const int b   = blockIdx.x >> 5;
  const int R0  = ((blockIdx.x & 31) << 7) + w * 16;   // this wave's row strip
  const float scale = 0.0625f;                          // 1/sqrt(256)

  // q A-fragments for the strip (K = 256 -> 8 fragments), from global bf16.
  Frag16 qf[8];
  const __bf16* qrow = qb + ((size_t)(b * N_DIM + R0 + m)) * C_DIM;
#pragma unroll
  for (int kk = 0; kk < 8; ++kk) {
    qf[kk].q[0] = *(const u32x4*)(qrow + kk * 32 + 16 * h);
    qf[kk].q[1] = *(const u32x4*)(qrow + kk * 32 + 16 * h + 8);
  }

  float e_r[8];
#pragma unroll
  for (int r = 0; r < 8; ++r) e_r[r] = e[R0 + r + 8 * h];

  f32x8 O[16];
#pragma unroll
  for (int ct = 0; ct < 16; ++ct)
#pragma unroll
    for (int i = 0; i < 8; ++i) O[ct][i] = 0.f;
  float lsum[8];
#pragma unroll
  for (int r = 0; r < 8; ++r) lsum[r] = 0.f;

  __bf16* mypb = &sp[w * 16 * PLD];

  // Prime the pipeline: stage step 0 into buffer 0.
  stage_kv(kb, vT, b, 0, tid, sk[0], sv[0]);

  for (int step = 0; step < NSTEP; ++step) {
    const int bufi = step & 1;
    // Prefetch next step into the other buffer (safe: the end-of-step barrier
    // of step-1 guarantees everyone is done reading it), then wait for the
    // current buffer's 8 oldest async loads while the new 8 stay in flight.
    if (step + 1 < NSTEP) {
      stage_kv(kb, vT, b, (step + 1) * 32, tid, sk[bufi ^ 1], sv[bufi ^ 1]);
      wait_async<8>();
    } else {
      wait_async<0>();
    }
    __syncthreads();

    const __bf16* skb = sk[bufi];
    const __bf16* svb = sv[bufi];
    const int col0 = step * 32;

    // ---- S = q @ k^T : two 16x16 tiles over K=256 ----
    f32x8 S0, S1;
#pragma unroll
    for (int i = 0; i < 8; ++i) { S0[i] = 0.f; S1[i] = 0.f; }
#pragma unroll
    for (int kk = 0; kk < 8; ++kk) {
      Frag16 b0, b1;
      const __bf16* kb0 = &skb[m * KLD + kk * 32 + 16 * h];
      const __bf16* kb1 = &skb[(m + 16) * KLD + kk * 32 + 16 * h];
      b0.q[0] = *(const u32x4*)kb0; b0.q[1] = *(const u32x4*)(kb0 + 8);
      b1.q[0] = *(const u32x4*)kb1; b1.q[1] = *(const u32x4*)(kb1 + 8);
      S0 = __builtin_amdgcn_wmma_f32_16x16x32_bf16(false, qf[kk].v, false, b0.v,
                                                   (short)0, S0, false, false);
      S1 = __builtin_amdgcn_wmma_f32_16x16x32_bf16(false, qf[kk].v, false, b1.v,
                                                   (short)0, S1, false, false);
    }

    // ---- clip, write H_hat, p = exp(clip - 10), row-sum ----
    float p0[8], p1[8];
    const size_t hbase =
        ((size_t)(b * N_DIM + R0 + 8 * h)) * N_DIM + col0 + m;
#pragma unroll
    for (int r = 0; r < 8; ++r) {
      float c0 = fminf(fmaxf(S0[r] * scale, -10.f), 10.f);
      float c1 = fminf(fmaxf(S1[r] * scale, -10.f), 10.f);
      Hout[hbase + (size_t)r * N_DIM]      = c0 + e_r[r];
      Hout[hbase + (size_t)r * N_DIM + 16] = c1 + e_r[r];
      p0[r] = __expf(c0 - 10.f);
      p1[r] = __expf(c1 - 10.f);
      lsum[r] += p0[r] + p1[r];
    }

    // ---- transpose P (D layout -> A layout) via per-wave LDS buffer ----
#pragma unroll
    for (int r = 0; r < 8; ++r) {
      mypb[(r + 8 * h) * PLD + m]      = (__bf16)p0[r];
      mypb[(r + 8 * h) * PLD + 16 + m] = (__bf16)p1[r];
    }
    __builtin_amdgcn_wave_barrier();
    Frag16 pf;
    pf.q[0] = *(const u32x4*)&mypb[m * PLD + 16 * h];
    pf.q[1] = *(const u32x4*)&mypb[m * PLD + 16 * h + 8];

    // ---- O += P @ V : 16 column tiles, K = 32 ----
#pragma unroll
    for (int ct = 0; ct < 16; ++ct) {
      Frag16 vf;
      const __bf16* vb = &svb[(ct * 16 + m) * VLD + 16 * h];
      vf.q[0] = *(const u32x4*)vb;
      vf.q[1] = *(const u32x4*)(vb + 8);
      O[ct] = __builtin_amdgcn_wmma_f32_16x16x32_bf16(false, pf.v, false, vf.v,
                                                      (short)0, O[ct], false, false);
    }
    __syncthreads();  // all waves done with sk/sv[bufi] -> reusable next step
  }

  // ---- epilogue: row sums across 16-lane groups, gate scaling, store ----
#pragma unroll
  for (int r = 0; r < 8; ++r) {
#pragma unroll
    for (int off = 8; off >= 1; off >>= 1)
      lsum[r] += __shfl_xor(lsum[r], off, 16);
  }
  float fs[8];
#pragma unroll
  for (int r = 0; r < 8; ++r) {
    float gv   = g[R0 + r + 8 * h];
    float gate = 1.f / (1.f + __expf(-gv));
    fs[r] = gate * log1pf(gate) / lsum[r];   // gate * log1p(gate) / softmax-denom
  }
#pragma unroll
  for (int ct = 0; ct < 16; ++ct) {
    const size_t obase =
        ((size_t)(b * N_DIM + R0 + 8 * h)) * C_DIM + ct * 16 + m;
#pragma unroll
    for (int r = 0; r < 8; ++r)
      Vout[obase + (size_t)r * C_DIM] = O[ct][r] * fs[r];
  }
}

// ---------------------------------------------------------------------------
extern "C" void kernel_launch(void* const* d_in, const int* in_sizes, int n_in,
                              void* d_out, int out_size, void* d_ws, size_t ws_size,
                              hipStream_t stream) {
  const float* nodes = (const float*)d_in[0];
  const float* g     = (const float*)d_in[1];
  const float* e     = (const float*)d_in[2];
  const float* Wq    = (const float*)d_in[3];
  const float* bq    = (const float*)d_in[4];
  const float* Wk    = (const float*)d_in[5];
  const float* bk    = (const float*)d_in[6];
  const float* Wv    = (const float*)d_in[7];
  const float* bv    = (const float*)d_in[8];

  char* ws = (char*)d_ws;
  const size_t WSZ  = (size_t)C_DIM * C_DIM * sizeof(__bf16);         // 128 KB
  const size_t QSZ  = (size_t)B_DIM * N_DIM * C_DIM * sizeof(__bf16); // 4 MB
  __bf16* wqb = (__bf16*)(ws);
  __bf16* wkb = (__bf16*)(ws + WSZ);
  __bf16* wvb = (__bf16*)(ws + 2 * WSZ);
  __bf16* qb  = (__bf16*)(ws + 3 * WSZ);
  __bf16* kb  = (__bf16*)(ws + 3 * WSZ + QSZ);
  __bf16* vT  = (__bf16*)(ws + 3 * WSZ + 2 * QSZ);

  float* Vout = (float*)d_out;
  float* Hout = Vout + (size_t)B_DIM * N_DIM * C_DIM;

  wcvt_kernel<<<768, 256, 0, stream>>>(Wq, Wk, Wv, wqb, wkb, wvb);
  qkv_kernel<<<B_DIM * (N_DIM / 64), 256, 0, stream>>>(
      nodes, wqb, wkb, wvb, bq, bk, bv, qb, kb, vT);
  attn_kernel<<<B_DIM * (N_DIM / 128), 256, 0, stream>>>(
      qb, kb, vT, g, e, Vout, Hout);
}
